// mixprop_attention_20787641712698
// MI455X (gfx1250) — matrix operand order repaired and checked by reading, not compiled, not run on hardware
//
#include <hip/hip_runtime.h>

#define BN 32
#define CIN 32
#define COUT 32
#define NNODE 207
#define LDIM 64
#define DKV 40
#define NPAD 224            // node dim padded for K (7*32)
#define ROWPAD 208          // node dim padded to 16 (13*16)
#define NSLICES 1024        // B * C
#define SLICE_IN (NNODE * LDIM)    // 13248
#define SLICE_PAD (NPAD * LDIM)    // 14336 (transposed: 64 rows x 224 cols)
#define AMIX 0.05f
#define LN_EPS 1e-6f

typedef __attribute__((ext_vector_type(16))) _Float16 v16h;
typedef __attribute__((ext_vector_type(8)))  float    v8f;

union u16h16 { uint4 q[2]; v16h v; };

__device__ __forceinline__ v8f vzero() {
  v8f z;
#pragma unroll
  for (int i = 0; i < 8; ++i) z[i] = 0.f;
  return z;
}

__device__ __forceinline__ v8f wmma_f16(v16h a, v16h b, v8f c) {
  return __builtin_amdgcn_wmma_f32_16x16x32_f16(false, a, false, b, (short)0, c,
                                                false, false);
}

// A operand 16x32 f16, row-major [m][k] with leading dim ld: per-lane 2x b128.
__device__ __forceinline__ v16h frag_a_ld(const _Float16* base, int ld) {
  int lane = threadIdx.x & 31;
  int m  = lane & 15;
  int kb = (lane & 16) ? 8 : 0;
  const _Float16* p = base + m * ld + kb;
  u16h16 u;
  u.q[0] = *(const uint4*)p;          // halves kb..kb+7
  u.q[1] = *(const uint4*)(p + 16);   // halves 16+kb..16+kb+7
  return u.v;
}
// B operand 32x16 f16 given B^T row-major [n][k] with leading dim ld: 2x b128.
__device__ __forceinline__ v16h frag_bt_ld(const _Float16* base, int ld) {
  int lane = threadIdx.x & 31;
  int n  = lane & 15;
  int kb = (lane & 16) ? 8 : 0;
  const _Float16* p = base + n * ld + kb;
  u16h16 u;
  u.q[0] = *(const uint4*)p;          // halves kb..kb+7
  u.q[1] = *(const uint4*)(p + 16);   // halves 16+kb..16+kb+7
  return u.v;
}
// C/D 16x16 f32 scatter: lanes 0-15 N=lane M=r, lanes 16-31 N=lane-16 M=8+r
template <typename F>
__device__ __forceinline__ void frag_c(F f, v8f c) {  // f(m,n,val)
  int lane = threadIdx.x & 31;
  int n  = lane & 15;
  int mb = (lane & 16) ? 8 : 0;
#pragma unroll
  for (int r = 0; r < 8; ++r) f(mb + r, n, c[r]);
}

// ------------- kernel 0: normalize adjacency, pad/convert all weights -------------
__global__ void k_prep(const float* __restrict__ adj, const float* __restrict__ wmlp,
                       const float* __restrict__ wq, const float* __restrict__ wk,
                       const float* __restrict__ wv, const float* __restrict__ wfc,
                       _Float16* __restrict__ a16, _Float16* __restrict__ wm16,
                       _Float16* __restrict__ wqp, _Float16* __restrict__ wkp,
                       _Float16* __restrict__ wvp, _Float16* __restrict__ wfcp) {
  int t = threadIdx.x;
  for (int i = t; i < ROWPAD * NPAD; i += blockDim.x) a16[i] = (_Float16)0.f;
  __syncthreads();
  for (int v = t; v < NNODE; v += blockDim.x) {
    float s = 1.0f;  // +I diagonal
    for (int w = 0; w < NNODE; ++w) s += adj[v * NNODE + w];
    float inv = 1.0f / s;
    for (int w = 0; w < NNODE; ++w) {
      float val = adj[v * NNODE + w] + (v == w ? 1.f : 0.f);
      a16[v * NPAD + w] = (_Float16)(val * inv);
    }
  }
  for (int i = t; i < COUT * 3 * CIN; i += blockDim.x) wm16[i] = (_Float16)wmlp[i];
  for (int i = t; i < 48 * LDIM; i += blockDim.x) {      // [d(48 pad)][l]
    int d = i / LDIM, l = i % LDIM;
    float q = (d < DKV) ? wq[d * LDIM + l] : 0.f;
    float k = (d < DKV) ? wk[d * LDIM + l] : 0.f;
    float v = (d < DKV) ? wv[d * LDIM + l] : 0.f;
    wqp[i] = (_Float16)q; wkp[i] = (_Float16)k; wvp[i] = (_Float16)v;
  }
  for (int i = t; i < LDIM * LDIM; i += blockDim.x) {    // [l][d(64 pad)]
    int l = i / LDIM, d = i % LDIM;
    wfcp[i] = (_Float16)((d < DKV) ? wfc[l * DKV + d] : 0.f);
  }
}

// ------------- kernel 1: x -> transposed f16 [l][w] padded, zero h buffers -------------
__global__ void k_init(const float* __restrict__ x, _Float16* __restrict__ xT,
                       _Float16* __restrict__ h1, _Float16* __restrict__ h2) {
  int i = blockIdx.x * blockDim.x + threadIdx.x;
  if (i >= NSLICES * SLICE_PAD) return;
  int slice = i / SLICE_PAD;
  int rem   = i % SLICE_PAD;
  int l = rem / NPAD, w = rem % NPAD;
  float v = (w < NNODE) ? x[(size_t)slice * SLICE_IN + w * LDIM + l] : 0.f;
  xT[i] = (_Float16)v;
  h1[i] = (_Float16)0.f;
  h2[i] = (_Float16)0.f;
}

// ------------- kernel 2: houtT[l][v] = alpha*xT + (1-alpha) * sum_w hT[l][w]*a[v][w] -------------
// One wave per (slice, 16-row l-tile). A = hT rows (contiguous), B^T = a16 rows (contiguous).
__global__ void __launch_bounds__(32)
k_nconv(const _Float16* __restrict__ a16,
        const _Float16* __restrict__ hinT,
        const _Float16* __restrict__ xT,
        _Float16* __restrict__ houtT) {
  __shared__ _Float16 ot[16 * ROWPAD];   // output tile [l 16][v 208]
  int blk = blockIdx.x;                  // NSLICES * 4
  int slice = blk >> 2;
  int m0 = (blk & 3) * 16;               // l-tile
  const _Float16* hsT = hinT + (size_t)slice * SLICE_PAD;
  const _Float16* xsT = xT + (size_t)slice * SLICE_PAD;
  _Float16* osT = houtT + (size_t)slice * SLICE_PAD;
  const int t = threadIdx.x;
  v8f acc[13];
#pragma unroll
  for (int i = 0; i < 13; ++i) acc[i] = vzero();
#pragma unroll 1
  for (int kk = 0; kk < 7; ++kk) {
    int k0 = kk * 32;
    if (kk < 6) __builtin_prefetch(hsT + m0 * NPAD + k0 + 32, 0, 1);
    v16h af = frag_a_ld(hsT + m0 * NPAD + k0, NPAD);
#pragma unroll
    for (int nt = 0; nt < 13; ++nt) {
      v16h bf = frag_bt_ld(a16 + nt * 16 * NPAD + k0, NPAD);
      acc[nt] = wmma_f16(af, bf, acc[nt]);
    }
  }
#pragma unroll
  for (int nt = 0; nt < 13; ++nt) {
    int n0 = nt * 16;
    frag_c([&](int m, int n, float v) {
      ot[m * ROWPAD + n0 + n] = (_Float16)((1.f - AMIX) * v);
    }, acc[nt]);
  }
  __syncthreads();
  // fuse alpha*xT and store rows m0..m0+15, cols 0..207 (cols 208..223 stay zero)
#pragma unroll
  for (int j = 0; j < 13; ++j) {
    int idx = t + j * 32;                // 416 uint4 = 16 rows x 26
    int r = idx / 26, c8 = (idx % 26) * 8;
    u16h16 xv, ov, res;
    xv.q[0] = *(const uint4*)(xsT + (m0 + r) * NPAD + c8);
    ov.q[0] = *(const uint4*)(ot + r * ROWPAD + c8);
#pragma unroll
    for (int e = 0; e < 8; ++e) {
      _Float16 a = ((const _Float16*)&xv.q[0])[e];
      _Float16 b = ((const _Float16*)&ov.q[0])[e];
      ((_Float16*)&res.q[0])[e] = (_Float16)(AMIX * (float)a + (float)b);
    }
    *(uint4*)(osT + (m0 + r) * NPAD + c8) = res.q[0];
  }
}

// ------------- kernel 3: s = w_mlp @ concat(x,h1,h2) + b  -> d_out (f32, [b][o][n][l]) -------------
// One wave per (b, l, 16-node tile); 1KB LDS transpose bounce for the channel-K operand.
__global__ void __launch_bounds__(32)
k_chanmix(const _Float16* __restrict__ wm16,
          const _Float16* __restrict__ xT,
          const _Float16* __restrict__ h1,
          const _Float16* __restrict__ h2,
          const float* __restrict__ bmlp,
          float* __restrict__ out) {
  __shared__ _Float16 bsT[16 * 32];      // [pos n'][channel k]
  int blk = blockIdx.x;                  // BN * LDIM * 13
  int nt = blk % 13;
  int l  = (blk / 13) % LDIM;
  int b  = blk / (13 * LDIM);
  int n0 = nt * 16;
  const _Float16* srcs[3] = {xT, h1, h2};
  const int t = threadIdx.x;
  v8f a0 = vzero(), a1 = vzero();
#pragma unroll 1
  for (int ch = 0; ch < 3; ++ch) {
    // thread t = channel t: load 16 contiguous halves (node n0..n0+15 at this l)
    const _Float16* src = srcs[ch] + ((size_t)(b * CIN + t)) * SLICE_PAD + l * NPAD + n0;
    u16h16 row;
    row.q[0] = *(const uint4*)src;
    row.q[1] = *(const uint4*)(src + 8);
#pragma unroll
    for (int j = 0; j < 16; ++j) bsT[j * 32 + t] = ((const _Float16*)&row)[j];
    __syncthreads();
    v16h bf  = frag_bt_ld(bsT, 32);
    v16h af0 = frag_a_ld(wm16 + ch * 32, 96);
    v16h af1 = frag_a_ld(wm16 + 16 * 96 + ch * 32, 96);
    a0 = wmma_f16(af0, bf, a0);
    a1 = wmma_f16(af1, bf, a1);
    __syncthreads();
  }
  float* ob = out + ((size_t)b * COUT * NNODE) * LDIM + l;
  frag_c([&](int m, int nn, float v) {
    ob[((size_t)m * NNODE + n0 + nn) * LDIM] = v + bmlp[m];
  }, a0);
  frag_c([&](int m, int nn, float v) {
    int o = m + 16;
    ob[((size_t)o * NNODE + n0 + nn) * LDIM] = v + bmlp[o];
  }, a1);
}

// ------------- kernel 4: per-slice attention + fc + residual + LN -------------
__global__ void __launch_bounds__(256)
k_attn(const _Float16* __restrict__ wqp, const _Float16* __restrict__ wkp,
       const _Float16* __restrict__ wvp, const _Float16* __restrict__ wfcp,
       const float* __restrict__ lng, const float* __restrict__ lnb,
       float* __restrict__ out) {
  __shared__ union {
    _Float16 s16[NPAD * LDIM];           // staged s (dead after q/k/v)
    float attnb[32 * NPAD];              // f32 logits / probs per row block
  } u;
  __shared__ _Float16 q16[NPAD][LDIM];   // q*invt, cols 40..63 zero
  __shared__ _Float16 k16[NPAD][LDIM];
  __shared__ _Float16 vT[48][NPAD];      // v transposed [d][p]
  __shared__ _Float16 attn16[32][NPAD];  // probs f16 (A operand of attn@v)
  __shared__ _Float16 ob16[32][LDIM];    // o, cols 48..63 zero
  __shared__ float fcb[32][LDIM];

  const int tid = threadIdx.x;
  const int wave = tid >> 5, lane = tid & 31;
  float* sg = out + (size_t)blockIdx.x * SLICE_IN;   // this slice's s / output

  for (int idx = tid; idx < NPAD * LDIM; idx += 256) {
    float v = (idx < SLICE_IN) ? sg[idx] : 0.f;      // rows = n, row-major [n][l]
    u.s16[idx] = (_Float16)v;
  }
  for (int i = tid; i < NPAD * 16; i += 256) {       // zero K-pad cols of q/k
    int r = i / 16, c = 48 + (i % 16);
    q16[r][c] = (_Float16)0.f;
    k16[r][c] = (_Float16)0.f;
  }
  for (int i = tid; i < 32 * LDIM; i += 256)         // ob16 cols 48..63 stay zero
    ob16[i / LDIM][i % LDIM] = (_Float16)0.f;
  __syncthreads();

  const float invt = rsqrtf((float)DKV);
  // q/k/v: A = s16 [n][l], B^T = padded weights [d][l]; 3*14*3 = 126 wave-jobs
  for (int job = wave; job < 126; job += 8) {
    int t = job / 42, rem = job % 42;
    int m0 = (rem / 3) * 16, n0 = (rem % 3) * 16;
    const _Float16* w = (t == 0) ? wqp : (t == 1) ? wkp : wvp;
    v8f acc = vzero();
#pragma unroll
    for (int kk = 0; kk < 2; ++kk) {
      int k0 = kk * 32;
      v16h af = frag_a_ld(u.s16 + m0 * LDIM + k0, LDIM);
      v16h bf = frag_bt_ld(w + n0 * LDIM + k0, LDIM);
      acc = wmma_f16(af, bf, acc);
    }
    if (t == 0)
      frag_c([&](int m, int n, float v) { q16[m0 + m][n0 + n] = (_Float16)(v * invt); }, acc);
    else if (t == 1)
      frag_c([&](int m, int n, float v) { k16[m0 + m][n0 + n] = (_Float16)v; }, acc);
    else  // v stored transposed: per-lane contiguous run of 8 halves
      frag_c([&](int m, int n, float v) { vT[n0 + n][m0 + m] = (_Float16)v; }, acc);
  }
  __syncthreads();

  for (int r0 = 0; r0 < NPAD; r0 += 32) {   // 7 row blocks of 32 query rows
    // logits = q @ k^T : both operands row-contiguous
    for (int job = wave; job < 28; job += 8) {
      int mt = job / 14, nt = job % 14;
      int m0 = r0 + mt * 16, n0 = nt * 16;
      v8f acc = vzero();
#pragma unroll
      for (int kk = 0; kk < 2; ++kk) {
        int k0 = kk * 32;
        v16h af = frag_a_ld(&q16[m0][k0], LDIM);
        v16h bf = frag_bt_ld(&k16[n0][k0], LDIM);
        acc = wmma_f16(af, bf, acc);
      }
      frag_c([&](int m, int n, float v) { u.attnb[(mt * 16 + m) * NPAD + n0 + n] = v; }, acc);
    }
    __syncthreads();

    // ELU + softmax over valid columns [0,207); write normalized probs as f16
    for (int r = wave; r < 32; r += 8) {
      float mx = -1e30f;
      for (int c = lane; c < NNODE; c += 32) {
        float xv = u.attnb[r * NPAD + c];
        float e = xv > 0.f ? xv : (__expf(xv) - 1.f);
        u.attnb[r * NPAD + c] = e;
        mx = fmaxf(mx, e);
      }
#pragma unroll
      for (int off = 16; off; off >>= 1) mx = fmaxf(mx, __shfl_xor(mx, off));
      float sm = 0.f;
      for (int c = lane; c < NPAD; c += 32) {
        float p = (c < NNODE) ? __expf(u.attnb[r * NPAD + c] - mx) : 0.f;
        u.attnb[r * NPAD + c] = p;
        sm += p;
      }
#pragma unroll
      for (int off = 16; off; off >>= 1) sm += __shfl_xor(sm, off);
      float inv = 1.f / sm;
      for (int c = lane; c < NPAD; c += 32)
        attn16[r][c] = (_Float16)(u.attnb[r * NPAD + c] * inv);
    }
    __syncthreads();

    // o = attn @ v : A = attn16 [n][p], B^T = vT [d][p]
    for (int job = wave; job < 6; job += 8) {
      int m0 = (job / 3) * 16, n0 = (job % 3) * 16;
      v8f acc = vzero();
#pragma unroll 1
      for (int kk = 0; kk < 7; ++kk) {
        int k0 = kk * 32;
        v16h af = frag_a_ld(&attn16[m0][k0], NPAD);
        v16h bf = frag_bt_ld(&vT[n0][k0], NPAD);
        acc = wmma_f16(af, bf, acc);
      }
      frag_c([&](int m, int n, float v) { ob16[m0 + m][n0 + n] = (_Float16)v; }, acc);
    }
    __syncthreads();

    // fc: A = ob16 [n][d], B^T = wfcp [l][d]
    for (int job = wave; job < 8; job += 8) {
      int m0 = (job / 4) * 16, n0 = (job % 4) * 16;
      v8f acc = vzero();
#pragma unroll
      for (int kk = 0; kk < 2; ++kk) {
        int k0 = kk * 32;
        v16h af = frag_a_ld(&ob16[m0][k0], LDIM);
        v16h bf = frag_bt_ld(wfcp + n0 * LDIM + k0, LDIM);
        acc = wmma_f16(af, bf, acc);
      }
      frag_c([&](int m, int n, float v) { fcb[m0 + m][n0 + n] = v; }, acc);
    }
    __syncthreads();

    // residual + LayerNorm over L=64, write back in place
    for (int r = wave; r < 32; r += 8) {
      int gr = r0 + r;
      if (gr < NNODE) {
        float v0 = fcb[r][lane]      + sg[gr * LDIM + lane];
        float v1 = fcb[r][lane + 32] + sg[gr * LDIM + lane + 32];
        float sum = v0 + v1;
#pragma unroll
        for (int off = 16; off; off >>= 1) sum += __shfl_xor(sum, off);
        float mu = sum * (1.f / 64.f);
        float d0 = v0 - mu, d1 = v1 - mu;
        float vs = d0 * d0 + d1 * d1;
#pragma unroll
        for (int off = 16; off; off >>= 1) vs += __shfl_xor(vs, off);
        float rstd = rsqrtf(vs * (1.f / 64.f) + LN_EPS);
        sg[gr * LDIM + lane]      = d0 * rstd * lng[lane]      + lnb[lane];
        sg[gr * LDIM + lane + 32] = d1 * rstd * lng[lane + 32] + lnb[lane + 32];
      }
    }
    __syncthreads();
  }
}

extern "C" void kernel_launch(void* const* d_in, const int* in_sizes, int n_in,
                              void* d_out, int out_size, void* d_ws, size_t ws_size,
                              hipStream_t stream) {
  (void)in_sizes; (void)n_in; (void)out_size; (void)ws_size;
  const float* x    = (const float*)d_in[0];
  const float* adj  = (const float*)d_in[1];
  const float* wmlp = (const float*)d_in[2];
  const float* bmlp = (const float*)d_in[3];
  const float* wq   = (const float*)d_in[4];
  const float* wk   = (const float*)d_in[5];
  const float* wv   = (const float*)d_in[6];
  const float* wfc  = (const float*)d_in[7];
  const float* lng  = (const float*)d_in[8];
  const float* lnb  = (const float*)d_in[9];
  float* out = (float*)d_out;

  // workspace (f16): xT | h1T | h2T | a16 | wm16 | wqp | wkp | wvp | wfcp (~88.3 MB)
  _Float16* xT   = (_Float16*)d_ws;
  _Float16* h1   = xT  + (size_t)NSLICES * SLICE_PAD;
  _Float16* h2   = h1  + (size_t)NSLICES * SLICE_PAD;
  _Float16* a16  = h2  + (size_t)NSLICES * SLICE_PAD;
  _Float16* wm16 = a16 + (size_t)ROWPAD * NPAD;
  _Float16* wqp  = wm16 + COUT * 3 * CIN;
  _Float16* wkp  = wqp + 48 * LDIM;
  _Float16* wvp  = wkp + 48 * LDIM;
  _Float16* wfcp = wvp + 48 * LDIM;

  k_prep<<<1, 256, 0, stream>>>(adj, wmlp, wq, wk, wv, wfc,
                                a16, wm16, wqp, wkp, wvp, wfcp);
  int tot = NSLICES * SLICE_PAD;
  k_init<<<(tot + 255) / 256, 256, 0, stream>>>(x, xT, h1, h2);
  k_nconv<<<NSLICES * 4, 32, 0, stream>>>(a16, xT, xT, h1);
  k_nconv<<<NSLICES * 4, 32, 0, stream>>>(a16, h1, xT, h2);
  k_chanmix<<<BN * LDIM * 13, 32, 0, stream>>>(wm16, xT, h1, h2, bmlp, out);
  k_attn<<<NSLICES, 256, 0, stream>>>(wqp, wkp, wvp, wfcp, lng, lnb, out);
}
